// EnhancedSwinTransformerBlock_90787018703087
// MI455X (gfx1250) — matrix operand order repaired and checked by reading, not compiled, not run on hardware
//
#include <hip/hip_runtime.h>

#define M_TOK 401408
#define TOK_PER_B 50176

typedef __attribute__((ext_vector_type(16))) __bf16 v16bf;
typedef __attribute__((ext_vector_type(8)))  float  v8f;

struct __align__(16) U4 { unsigned int a, b, c, d; };

union FragB16 {
  v16bf v;
  U4 q[2];
  unsigned short h[16];
};

__device__ __forceinline__ unsigned short f2bf(float f) {
  unsigned int u = __float_as_uint(f);
  u += 0x7FFFu + ((u >> 16) & 1u);          // round-to-nearest-even
  return (unsigned short)(u >> 16);
}
__device__ __forceinline__ float bf2f(unsigned short h) {
  return __uint_as_float(((unsigned int)h) << 16);
}
__device__ __forceinline__ v8f zero8() {
  v8f z = {0.f, 0.f, 0.f, 0.f, 0.f, 0.f, 0.f, 0.f};
  return z;
}

__device__ __forceinline__ v8f wmma_bf16(const FragB16& a, const FragB16& b, v8f c) {
  return __builtin_amdgcn_wmma_f32_16x16x32_bf16(false, a.v, false, b.v, (short)0, c,
                                                 false, false);
}

// A-matrix 16x32 bf16 fragment from row-major [rows][ld] (global or LDS).
// lanes 0-15: K = 0..15; lanes 16-31: K = 16..31 (contiguous, matches B fragments).
__device__ __forceinline__ void load_a(FragB16& f, const unsigned short* base, int ld,
                                       int m0, int k0) {
  int l = threadIdx.x & 31;
  const unsigned short* row = base + (size_t)(m0 + (l & 15)) * ld;
  int klo = (l >> 4) << 4;
  f.q[0] = *(const U4*)(row + k0 + klo);
  f.q[1] = *(const U4*)(row + k0 + klo + 8);
}

// B-matrix 32x16 fragment from pre-swizzled weight buffer: 2 contiguous b128 loads.
__device__ __forceinline__ void load_bw(FragB16& f, const unsigned short* wsw, int NT,
                                        int kt, int nt) {
  int l = threadIdx.x & 31;
  const unsigned short* p = wsw + (((size_t)(kt * NT + nt) * 32 + l) << 4);
  f.q[0] = *(const U4*)p;
  f.q[1] = *(const U4*)(p + 8);
}

// B-matrix 32x16 fragment where memory holds B column-major: mem[n][k], stride ld.
// lanes 0-15 take K=0..15, lanes 16-31 take K=16..31, contiguous in k -> b128 loads.
__device__ __forceinline__ void load_b_cm(FragB16& f, const unsigned short* bcm, int ld,
                                          int k0, int n0) {
  int l = threadIdx.x & 31;
  const unsigned short* p = bcm + (size_t)(n0 + (l & 15)) * ld + k0 + ((l >> 4) << 4);
  f.q[0] = *(const U4*)p;
  f.q[1] = *(const U4*)(p + 8);
}

// ---------------- weight prep: fp32 -> bf16, fragment-swizzled layout ---------------
__device__ void swz(unsigned short* dst, const float* src, int K, int N, int tid,
                    int nth) {
  int NT = N >> 4;
  int total = K * N;
  for (int i = tid; i < total; i += nth) {
    int t = i & 15;
    int lane = (i >> 4) & 31;
    int rem = i >> 9;
    int nt = rem % NT;
    int kt = rem / NT;
    int k = kt * 32 + ((lane >> 4) << 4) + t;
    int n = nt * 16 + (lane & 15);
    dst[i] = f2bf(src[(size_t)k * N + n]);
  }
}

__global__ void __launch_bounds__(256)
k_prep(const float* __restrict__ w_qkv, const float* __restrict__ w_proj,
       const float* __restrict__ w_gate, const float* __restrict__ w1,
       const float* __restrict__ w2, unsigned short* __restrict__ sq,
       unsigned short* __restrict__ spg, unsigned short* __restrict__ s1,
       unsigned short* __restrict__ s2, float* __restrict__ pool) {
  int tid = blockIdx.x * blockDim.x + threadIdx.x;
  int nth = gridDim.x * blockDim.x;
  swz(sq, w_qkv, 96, 288, tid, nth);
  {  // proj||gate concatenated to N=192
    int NT = 12, total = 96 * 192;
    for (int i = tid; i < total; i += nth) {
      int t = i & 15;
      int lane = (i >> 4) & 31;
      int rem = i >> 9;
      int nt = rem % NT;
      int kt = rem / NT;
      int k = kt * 32 + ((lane >> 4) << 4) + t;
      int n = nt * 16 + (lane & 15);
      float s = (n < 96) ? w_proj[k * 96 + n] : w_gate[k * 96 + (n - 96)];
      spg[i] = f2bf(s);
    }
  }
  swz(s1, w1, 96, 384, tid, nth);
  swz(s2, w2, 384, 96, tid, nth);
  for (int i = tid; i < 768; i += nth) pool[i] = 0.f;   // zero pooling accumulator
}

// ---------------- layernorm: one wave per token (96 channels) ----------------------
__global__ void __launch_bounds__(256)
k_ln(const float* __restrict__ xin, const float* __restrict__ g,
     const float* __restrict__ bb, unsigned short* __restrict__ yout) {
  int wave = threadIdx.x >> 5, lane = threadIdx.x & 31;
  size_t tok = (size_t)blockIdx.x * 8 + wave;
  const float* p = xin + tok * 96;
  float v0 = p[lane], v1 = p[lane + 32], v2 = p[lane + 64];
  float s = v0 + v1 + v2;
  float q = v0 * v0 + v1 * v1 + v2 * v2;
  for (int o = 16; o; o >>= 1) {
    s += __shfl_xor(s, o, 32);
    q += __shfl_xor(q, o, 32);
  }
  float mu = s * (1.f / 96.f);
  float var = q * (1.f / 96.f) - mu * mu;
  float rs = rsqrtf(var + 1e-5f);
  unsigned short* yo = yout + tok * 96;
  yo[lane]      = f2bf((v0 - mu) * rs * g[lane]      + bb[lane]);
  yo[lane + 32] = f2bf((v1 - mu) * rs * g[lane + 32] + bb[lane + 32]);
  yo[lane + 64] = f2bf((v2 - mu) * rs * g[lane + 64] + bb[lane + 64]);
}

// ---------------- QKV GEMM: [M,96] x [96,288] + bias, fold 1/sqrt(d) into Q --------
__global__ void __launch_bounds__(256)
k_gemm_qkv(const unsigned short* __restrict__ yln, const unsigned short* __restrict__ wsw,
           const float* __restrict__ bias, unsigned short* __restrict__ qkv) {
  const float scale = 0.17677669529663687f;  // 1/sqrt(32)
  int wave = threadIdx.x >> 5, lane = threadIdx.x & 31;
  int m0 = (blockIdx.x * 8 + wave) * 16;
  FragB16 fa[3];
#pragma unroll
  for (int ks = 0; ks < 3; ++ks) load_a(fa[ks], yln, 96, m0, ks * 32);
  for (int nc = 0; nc < 9; ++nc) {
    v8f a0 = zero8(), a1 = zero8();
#pragma unroll
    for (int ks = 0; ks < 3; ++ks) {
      FragB16 b0, b1;
      load_bw(b0, wsw, 18, ks, nc * 2);
      load_bw(b1, wsw, 18, ks, nc * 2 + 1);
      a0 = wmma_bf16(fa[ks], b0, a0);
      a1 = wmma_bf16(fa[ks], b1, a1);
    }
    int n = nc * 32 + (lane & 15);
    int mb = m0 + ((lane >> 4) << 3);
    float s0 = (n < 96) ? scale : 1.f;
    float s1v = (n + 16 < 96) ? scale : 1.f;
    float bb0 = bias[n], bb1 = bias[n + 16];
#pragma unroll
    for (int j = 0; j < 8; ++j) {
      qkv[(size_t)(mb + j) * 288 + n]      = f2bf((a0[j] + bb0) * s0);
      qkv[(size_t)(mb + j) * 288 + n + 16] = f2bf((a1[j] + bb1) * s1v);
    }
  }
}

// ---------------- windowed attention: one wave per (window, head) ------------------
__global__ void __launch_bounds__(32)
k_attn(const unsigned short* __restrict__ qkv, unsigned short* __restrict__ obf) {
  __shared__ __align__(16) unsigned short Qs[64 * 32];
  __shared__ __align__(16) unsigned short Ks[64 * 32];
  __shared__ __align__(16) unsigned short Vt[32 * 64];   // V transposed [dim][token]
  __shared__ __align__(16) unsigned short Ps[64 * 64];
  __shared__ __align__(16) float Ss[64 * 64];

  int wi = blockIdx.x / 3;
  int h = blockIdx.x - wi * 3;
  int b = wi >> 10;        // 32*32 windows per image
  int r = wi & 1023;
  int wh = r >> 5, ww = r & 31;
  int lane = threadIdx.x;

  __builtin_prefetch(qkv + (((size_t)b * 224 + wh * 7) * 224 + ww * 7) * 288, 0, 1);

  for (int i = lane; i < 64 * 32; i += 32) {
    int t = i >> 5, d = i & 31;
    unsigned short q = 0, k = 0, v = 0;
    if (t < 49) {
      int hr = wh * 7 + t / 7;
      int wc = ww * 7 + t % 7;
      size_t g = (((size_t)b * 224 + hr) * 224 + wc) * 288 + h * 32 + d;
      q = qkv[g];
      k = qkv[g + 96];
      v = qkv[g + 192];
    }
    Qs[t * 32 + d] = q;
    Ks[t * 32 + d] = k;
    Vt[d * 64 + t] = v;
  }

  // S = (Q*scale) @ K^T : 64x64 tile, K-dim 32
  for (int mt = 0; mt < 4; ++mt) {
    FragB16 fa;
    load_a(fa, Qs, 32, mt * 16, 0);
    for (int nt = 0; nt < 4; ++nt) {
      FragB16 fb;
      load_b_cm(fb, Ks, 32, 0, nt * 16);
      v8f acc = zero8();
      acc = wmma_bf16(fa, fb, acc);
      int n = nt * 16 + (lane & 15);
      int mb = mt * 16 + ((lane >> 4) << 3);
#pragma unroll
      for (int j = 0; j < 8; ++j) Ss[(mb + j) * 64 + n] = acc[j];
    }
  }

  // softmax over valid 49 cols, zero the padding -> P (bf16)
  for (int rr = lane; rr < 64; rr += 32) {
    if (rr < 49) {
      float mx = -3.0e38f;
      for (int c = 0; c < 49; ++c) mx = fmaxf(mx, Ss[rr * 64 + c]);
      float s = 0.f;
      for (int c = 0; c < 49; ++c) {
        float e = __expf(Ss[rr * 64 + c] - mx);
        Ss[rr * 64 + c] = e;
        s += e;
      }
      float inv = 1.f / s;
      for (int c = 0; c < 64; ++c)
        Ps[rr * 64 + c] = (c < 49) ? f2bf(Ss[rr * 64 + c] * inv) : (unsigned short)0;
    } else {
      for (int c = 0; c < 64; ++c) Ps[rr * 64 + c] = 0;
    }
  }

  // O = P @ V : 64x32, K-dim 64 (2 steps)
  for (int mt = 0; mt < 4; ++mt) {
    FragB16 fa0, fa1;
    load_a(fa0, Ps, 64, mt * 16, 0);
    load_a(fa1, Ps, 64, mt * 16, 32);
    for (int nt = 0; nt < 2; ++nt) {
      v8f acc = zero8();
      FragB16 fb0, fb1;
      load_b_cm(fb0, Vt, 64, 0, nt * 16);
      acc = wmma_bf16(fa0, fb0, acc);
      load_b_cm(fb1, Vt, 64, 32, nt * 16);
      acc = wmma_bf16(fa1, fb1, acc);
      int n = nt * 16 + (lane & 15);
      int mb = mt * 16 + ((lane >> 4) << 3);
#pragma unroll
      for (int j = 0; j < 8; ++j) {
        int m = mb + j;
        if (m < 49) {
          int hr = wh * 7 + m / 7;
          int wc = ww * 7 + m % 7;
          obf[(((size_t)b * 224 + hr) * 224 + wc) * 96 + h * 32 + n] = f2bf(acc[j]);
        }
      }
    }
  }
}

// ---------------- proj + sigmoid gate + residual: [M,96] x [96,192] ----------------
__global__ void __launch_bounds__(256)
k_projgate(const unsigned short* __restrict__ obf, const unsigned short* __restrict__ wsw,
           const float* __restrict__ x, float* __restrict__ x1) {
  int wave = threadIdx.x >> 5, lane = threadIdx.x & 31;
  int m0 = (blockIdx.x * 8 + wave) * 16;
  FragB16 fa[3];
#pragma unroll
  for (int ks = 0; ks < 3; ++ks) load_a(fa[ks], obf, 96, m0, ks * 32);
  for (int nc = 0; nc < 3; ++nc) {
    v8f p0 = zero8(), p1 = zero8(), g0 = zero8(), g1 = zero8();
#pragma unroll
    for (int ks = 0; ks < 3; ++ks) {
      FragB16 bp0, bp1, bg0, bg1;
      load_bw(bp0, wsw, 12, ks, nc * 2);
      load_bw(bp1, wsw, 12, ks, nc * 2 + 1);
      load_bw(bg0, wsw, 12, ks, 6 + nc * 2);
      load_bw(bg1, wsw, 12, ks, 6 + nc * 2 + 1);
      p0 = wmma_bf16(fa[ks], bp0, p0);
      p1 = wmma_bf16(fa[ks], bp1, p1);
      g0 = wmma_bf16(fa[ks], bg0, g0);
      g1 = wmma_bf16(fa[ks], bg1, g1);
    }
    int n = nc * 32 + (lane & 15);
    int mb = m0 + ((lane >> 4) << 3);
#pragma unroll
    for (int j = 0; j < 8; ++j) {
      size_t i0 = (size_t)(mb + j) * 96 + n;
      size_t i1 = i0 + 16;
      x1[i0] = x[i0] + p0[j] * (1.f / (1.f + __expf(-g0[j])));
      x1[i1] = x[i1] + p1[j] * (1.f / (1.f + __expf(-g1[j])));
    }
  }
}

// ---------------- MLP fc1 + exact GELU: [M,96] x [96,384] --------------------------
__global__ void __launch_bounds__(256)
k_mlp1(const unsigned short* __restrict__ y2, const unsigned short* __restrict__ wsw,
       const float* __restrict__ bias, unsigned short* __restrict__ h1) {
  int wave = threadIdx.x >> 5, lane = threadIdx.x & 31;
  int m0 = (blockIdx.x * 8 + wave) * 16;
  FragB16 fa[3];
#pragma unroll
  for (int ks = 0; ks < 3; ++ks) load_a(fa[ks], y2, 96, m0, ks * 32);
  for (int nc = 0; nc < 12; ++nc) {
    v8f a0 = zero8(), a1 = zero8();
#pragma unroll
    for (int ks = 0; ks < 3; ++ks) {
      FragB16 b0, b1;
      load_bw(b0, wsw, 24, ks, nc * 2);
      load_bw(b1, wsw, 24, ks, nc * 2 + 1);
      a0 = wmma_bf16(fa[ks], b0, a0);
      a1 = wmma_bf16(fa[ks], b1, a1);
    }
    int n = nc * 32 + (lane & 15);
    int mb = m0 + ((lane >> 4) << 3);
    float bb0 = bias[n], bb1 = bias[n + 16];
#pragma unroll
    for (int j = 0; j < 8; ++j) {
      float v0 = a0[j] + bb0;
      float v1 = a1[j] + bb1;
      v0 = 0.5f * v0 * (1.f + erff(v0 * 0.70710678118f));
      v1 = 0.5f * v1 * (1.f + erff(v1 * 0.70710678118f));
      h1[(size_t)(mb + j) * 384 + n] = f2bf(v0);
      h1[(size_t)(mb + j) * 384 + n + 16] = f2bf(v1);
    }
  }
}

// ---------------- MLP fc2 + fused global-average-pool accumulation -----------------
__global__ void __launch_bounds__(256)
k_mlp2(const unsigned short* __restrict__ h1, const unsigned short* __restrict__ wsw,
       const float* __restrict__ bias, unsigned short* __restrict__ ybuf,
       float* __restrict__ pool) {
  __shared__ float lpool[96];
  int tid = threadIdx.x;
  if (tid < 96) lpool[tid] = 0.f;
  __syncthreads();
  int wave = tid >> 5, lane = tid & 31;
  int m0 = (blockIdx.x * 8 + wave) * 16;
  FragB16 fa[12];
#pragma unroll
  for (int ks = 0; ks < 12; ++ks) load_a(fa[ks], h1, 384, m0, ks * 32);
  for (int nc = 0; nc < 3; ++nc) {
    v8f a0 = zero8(), a1 = zero8();
#pragma unroll
    for (int ks = 0; ks < 12; ++ks) {
      FragB16 b0, b1;
      load_bw(b0, wsw, 6, ks, nc * 2);
      load_bw(b1, wsw, 6, ks, nc * 2 + 1);
      a0 = wmma_bf16(fa[ks], b0, a0);
      a1 = wmma_bf16(fa[ks], b1, a1);
    }
    int n = nc * 32 + (lane & 15);
    int mb = m0 + ((lane >> 4) << 3);
    float bb0 = bias[n], bb1 = bias[n + 16];
#pragma unroll
    for (int j = 0; j < 8; ++j) {
      float v0 = a0[j] + bb0;
      float v1 = a1[j] + bb1;
      ybuf[(size_t)(mb + j) * 96 + n] = f2bf(v0);
      ybuf[(size_t)(mb + j) * 96 + n + 16] = f2bf(v1);
      atomicAdd(&lpool[n], v0);
      atomicAdd(&lpool[n + 16], v1);
    }
  }
  __syncthreads();
  if (tid < 96) {
    int b = (blockIdx.x * 128) / TOK_PER_B;   // 50176 % 128 == 0, never crosses batch
    atomicAdd(&pool[b * 96 + tid], lpool[tid]);
  }
}

// ---------------- SE: tiny, one block ----------------------------------------------
__global__ void __launch_bounds__(256)
k_se(const float* __restrict__ pool, const float* __restrict__ w1,
     const float* __restrict__ b1, const float* __restrict__ w2,
     const float* __restrict__ b2, float* __restrict__ se) {
  __shared__ float pbuf[8 * 96];
  __shared__ float s1[8 * 24];
  int tid = threadIdx.x;
  for (int i = tid; i < 768; i += 256) pbuf[i] = pool[i] * (1.f / (float)TOK_PER_B);
  __syncthreads();
  if (tid < 192) {
    int bb = tid / 24, j = tid % 24;
    float s = b1[j];
    for (int c = 0; c < 96; ++c) s += pbuf[bb * 96 + c] * w1[c * 24 + j];
    s1[tid] = s / (1.f + __expf(-s));  // swish
  }
  __syncthreads();
  for (int i = tid; i < 768; i += 256) {
    int bb = i / 96, c = i % 96;
    float s = b2[c];
    for (int j = 0; j < 24; ++j) s += s1[bb * 24 + j] * w2[j * 96 + c];
    se[i] = 1.f / (1.f + __expf(-s));
  }
}

// ---------------- final: out = x1 + y * se[b,c] ------------------------------------
__global__ void __launch_bounds__(256)
k_final(const float* __restrict__ x1, const unsigned short* __restrict__ ybuf,
        const float* __restrict__ se, float* __restrict__ out) {
  size_t total = (size_t)M_TOK * 96;
  for (size_t i = (size_t)blockIdx.x * blockDim.x + threadIdx.x; i < total;
       i += (size_t)gridDim.x * blockDim.x) {
    int c = (int)(i % 96);
    int b = (int)(i / ((size_t)TOK_PER_B * 96));
    out[i] = x1[i] + bf2f(ybuf[i]) * se[b * 96 + c];
  }
}

extern "C" void kernel_launch(void* const* d_in, const int* in_sizes, int n_in,
                              void* d_out, int out_size, void* d_ws, size_t ws_size,
                              hipStream_t stream) {
  (void)in_sizes; (void)n_in; (void)out_size; (void)ws_size;
  const float* x      = (const float*)d_in[0];
  const float* ln1_g  = (const float*)d_in[1];
  const float* ln1_b  = (const float*)d_in[2];
  const float* w_qkv  = (const float*)d_in[3];
  const float* b_qkv  = (const float*)d_in[4];
  const float* w_proj = (const float*)d_in[5];
  const float* w_gate = (const float*)d_in[6];
  const float* ln2_g  = (const float*)d_in[7];
  const float* ln2_b  = (const float*)d_in[8];
  const float* mlp_w1 = (const float*)d_in[9];
  const float* mlp_b1 = (const float*)d_in[10];
  const float* mlp_w2 = (const float*)d_in[11];
  const float* mlp_b2 = (const float*)d_in[12];
  const float* se_w1  = (const float*)d_in[13];
  const float* se_b1  = (const float*)d_in[14];
  const float* se_w2  = (const float*)d_in[15];
  const float* se_b2  = (const float*)d_in[16];

  char* ws = (char*)d_ws;
  unsigned short* wqkv = (unsigned short*)(ws + 0);                 // 96x288 swizzled
  unsigned short* wpg  = (unsigned short*)(ws + 55296);             // 96x192 swizzled
  unsigned short* w1s  = (unsigned short*)(ws + 92160);             // 96x384 swizzled
  unsigned short* w2s  = (unsigned short*)(ws + 165888);            // 384x96 swizzled
  float* pool          = (float*)(ws + 239616);                     // [8,96]
  float* sebuf         = (float*)(ws + 242688);                     // [8,96]
  unsigned short* yln  = (unsigned short*)(ws + 245760ull);         // A: [M,96] bf16
  unsigned short* qkv  = (unsigned short*)(ws + 77316096ull);       // B: [M,288] bf16
  unsigned short* obf  = (unsigned short*)(ws + 308527104ull);      // C: [M,96] bf16
  float* x1            = (float*)(ws + 385597440ull);               // D: [M,96] f32
  unsigned short* h1   = qkv;   // reuse B..C for [M,384] bf16 after attention
  unsigned short* ybuf = yln;   // reuse A for [M,96] bf16 after MLP1 consumed it
  float* out = (float*)d_out;

  k_prep<<<256, 256, 0, stream>>>(w_qkv, w_proj, w_gate, mlp_w1, mlp_w2, wqkv, wpg,
                                  w1s, w2s, pool);
  k_ln<<<M_TOK / 8, 256, 0, stream>>>(x, ln1_g, ln1_b, yln);
  k_gemm_qkv<<<M_TOK / 128, 256, 0, stream>>>(yln, wqkv, b_qkv, qkv);
  k_attn<<<8192 * 3, 32, 0, stream>>>(qkv, obf);
  k_projgate<<<M_TOK / 128, 256, 0, stream>>>(obf, wpg, x, x1);
  k_ln<<<M_TOK / 8, 256, 0, stream>>>(x1, ln2_g, ln2_b, yln);
  k_mlp1<<<M_TOK / 128, 256, 0, stream>>>(yln, w1s, mlp_b1, h1);
  k_mlp2<<<M_TOK / 128, 256, 0, stream>>>(h1, w2s, mlp_b2, ybuf, pool);
  k_se<<<1, 256, 0, stream>>>(pool, se_w1, se_b1, se_w2, se_b2, sebuf);
  k_final<<<8192, 256, 0, stream>>>(x1, ybuf, sebuf, out);
}
